// DepthWiseSeparableConv_1357209666008
// MI455X (gfx1250) — compile-verified, matrix operand order checked by reading, not run on hardware
//
#include <hip/hip_runtime.h>
#include <hip/hip_bf16.h>

typedef __attribute__((ext_vector_type(16))) _Float16 v16h;
typedef __attribute__((ext_vector_type(8)))  float    v8f;

#define CN   256
#define CO_  512
#define HH   56
#define WW   56
#define HW   (HH*WW)     // 3136
#define NB   32

// gfx1250 async global->LDS copy: 16B per lane, no VGPR round-trip (ASYNCcnt).
__device__ __forceinline__ void async_ld16(unsigned int lds_byte_off, const void* gaddr) {
    asm volatile("global_load_async_to_lds_b128 %0, %1, off"
                 :: "v"(lds_byte_off), "v"(gaddr) : "memory");
}
__device__ __forceinline__ void wait_async0() {
    asm volatile("s_wait_asynccnt 0x0" ::: "memory");
}

// ---------------------------------------------------------------------------
// Kernel 0: fold bn2 scale (+ pointwise bias) into fp16 weights, per-o const.
// ---------------------------------------------------------------------------
__global__ void prep_weights(const float* __restrict__ w_pw, const float* __restrict__ b_pw,
                             const float* __restrict__ g2,  const float* __restrict__ be2,
                             const float* __restrict__ mu2, const float* __restrict__ var2,
                             _Float16* __restrict__ w2f, float* __restrict__ const2) {
    const int o = blockIdx.x;
    const int c = threadIdx.x;
    const float s = g2[o] * rsqrtf(var2[o] + 1e-5f);
    w2f[o * CN + c] = (_Float16)(w_pw[o * CN + c] * s);
    if (c == 0) const2[o] = be2[o] - mu2[o] * s + b_pw[o] * s;
}

// ---------------------------------------------------------------------------
// Kernel 1: depthwise 3x3 (shared kernel) + bn1 + ReLU, NCHW fp32 -> [n][hw][c]
// fp16 via LDS transpose so GEMM A-tiles are K-contiguous.
// ---------------------------------------------------------------------------
__global__ void __launch_bounds__(224)
dwconv_bn_relu(const float* __restrict__ x, const float* __restrict__ w_dw,
               const float* __restrict__ b_dw,
               const float* __restrict__ g1, const float* __restrict__ be1,
               const float* __restrict__ mu1, const float* __restrict__ var1,
               _Float16* __restrict__ yt) {
    __shared__ _Float16 tile[32][60];   // [c_local][w], padded

    const int h  = blockIdx.x;          // 0..55
    const int cb = blockIdx.y * 32;     // channel block base
    const int n  = blockIdx.z;          // 0..31
    const int t  = threadIdx.x;         // 0..223
    const int w  = t % 56;
    const int ci = t / 56;              // 0..3

    const float k00 = w_dw[0], k01 = w_dw[1], k02 = w_dw[2];
    const float k10 = w_dw[3], k11 = w_dw[4], k12 = w_dw[5];
    const float k20 = w_dw[6], k21 = w_dw[7], k22 = w_dw[8];
    const float bd = b_dw[0];

    for (int it = 0; it < 8; ++it) {
        const int cl = ci * 8 + it;
        const int c  = cb + cl;
        const float* xp = x + (((long)n * CN + c) * HH + h) * WW;
        float acc = 0.f;
        if (h > 0) {
            const float* r = xp - WW;
            if (w > 0)  acc += k00 * r[w - 1];
            acc += k01 * r[w];
            if (w < 55) acc += k02 * r[w + 1];
        }
        {
            const float* r = xp;
            if (w > 0)  acc += k10 * r[w - 1];
            acc += k11 * r[w];
            if (w < 55) acc += k12 * r[w + 1];
        }
        if (h < 55) {
            const float* r = xp + WW;
            if (w > 0)  acc += k20 * r[w - 1];
            acc += k21 * r[w];
            if (w < 55) acc += k22 * r[w + 1];
        }
        const float s  = g1[c] * rsqrtf(var1[c] + 1e-5f);
        const float sh = be1[c] - mu1[c] * s + bd * s;
        const float v  = fmaxf(acc * s + sh, 0.f);
        tile[cl][w] = (_Float16)v;
    }
    __syncthreads();

    // coalesced transpose-out: yt[(n*HW + h*56 + w)*256 + cb + c], 16B per thread
    const int wo = t >> 2;            // 0..55
    const int cg = (t & 3) * 8;       // 0,8,16,24
    union { _Float16 h8[8]; uint4 q; } pk;
#pragma unroll
    for (int j = 0; j < 8; ++j) pk.h8[j] = tile[cg + j][wo];
    const long m = (long)n * HW + h * 56 + wo;
    *(uint4*)(yt + m * CN + cb + cg) = pk.q;
}

// ---------------------------------------------------------------------------
// Kernel 2: pointwise GEMM (M=n*hw, K=c, N=o) with v_wmma_f32_16x16x32_f16.
// 64x64 C-tile per 128-thread WG (4 waves); K-loop 8 x 32.
// Double-buffered LDS, filled by async global->LDS copies (ASYNCcnt) so the
// K+1 tile loads overlap the K-step WMMAs; one barrier per K-step.
// bn2(+bias) folded; ReLU; LDS-transpose epilogue -> coalesced NCHW stores.
// ---------------------------------------------------------------------------
__global__ void __launch_bounds__(128)
pwgemm_bn_relu(const _Float16* __restrict__ yt, const _Float16* __restrict__ w2f,
               const float* __restrict__ const2, float* __restrict__ z) {
    // two buffers of (A: 64x40 halfs = 5120 B) + (B: 64x40 halfs = 5120 B)
    // epilogue aliases the same block as [64][68] f32 = 17408 B
    __shared__ __align__(16) unsigned char smem[20480];

    const int mt  = blockIdx.x;            // 0..48  (3136/64)
    const int nt  = blockIdx.y;            // 0..7   (512/64)
    const int n   = blockIdx.z;            // 0..31
    const int hw0 = mt * 64;
    const int ob  = nt * 64;
    const int t    = threadIdx.x;
    const int lane = t & 31;
    const int wave = t >> 5;               // 0..3 -> M rows [wave*16, wave*16+16)
    const int l15  = lane & 15;
    const int khalf = (lane < 16) ? 0 : 8; // 16-bit A/B lane layout K offset

    const long mbase = (long)n * HW + hw0;

    // per-thread staging chunks: 256 chunks of 16B per matrix, 2 per thread
    const int r0 = t >> 2,          p0 = t & 3;
    const int r1 = (t + 128) >> 2,  p1 = (t + 128) & 3;
    const _Float16* gA0 = yt  + (mbase + r0) * CN + p0 * 8;
    const _Float16* gA1 = yt  + (mbase + r1) * CN + p1 * 8;
    const _Float16* gB0 = w2f + (long)(ob + r0) * CN + p0 * 8;
    const _Float16* gB1 = w2f + (long)(ob + r1) * CN + p1 * 8;
    const unsigned int lA0 = (r0 * 40 + p0 * 8) * 2;
    const unsigned int lA1 = (r1 * 40 + p1 * 8) * 2;
    const unsigned int lB0 = 5120 + (r0 * 40 + p0 * 8) * 2;
    const unsigned int lB1 = 5120 + (r1 * 40 + p1 * 8) * 2;

    v8f acc[4] = {};

    // prologue: async-fill buffer 0 with K-step 0
    async_ld16(lA0, gA0);
    async_ld16(lA1, gA1);
    async_ld16(lB0, gB0);
    async_ld16(lB1, gB1);
    wait_async0();
    __syncthreads();

    for (int kk = 0; kk < 8; ++kk) {
        const unsigned int buf = (unsigned int)(kk & 1) * 10240u;

        // prefetch next K-step into the other buffer (overlaps WMMAs below)
        if (kk < 7) {
            const int koff = (kk + 1) * 32;
            const unsigned int nb = (unsigned int)((kk + 1) & 1) * 10240u;
            async_ld16(nb + lA0, gA0 + koff);
            async_ld16(nb + lA1, gA1 + koff);
            async_ld16(nb + lB0, gB0 + koff);
            async_ld16(nb + lB1, gB1 + koff);
        }

        const _Float16* As = (const _Float16*)(smem + buf);
        const _Float16* Bs = As + 64 * 40;

        // A fragment: 16x32 fp16, lanes 0-15 K{0-7,16-23}, lanes 16-31 K{8-15,24-31}
        union { v16h v; uint4 q[2]; } af;
        const int arow = wave * 16 + l15;
        af.q[0] = *(const uint4*)(As + arow * 40 + khalf);
        af.q[1] = *(const uint4*)(As + arow * 40 + 16 + khalf);

#pragma unroll
        for (int bt = 0; bt < 4; ++bt) {
            union { v16h v; uint4 q[2]; } bf;
            const int brow = bt * 16 + l15;     // N column, K-contiguous in LDS
            bf.q[0] = *(const uint4*)(Bs + brow * 40 + khalf);
            bf.q[1] = *(const uint4*)(Bs + brow * 40 + 16 + khalf);
            acc[bt] = __builtin_amdgcn_wmma_f32_16x16x32_f16(
                false, af.v, false, bf.v, (short)0, acc[bt], false, false);
        }

        // publish: own prefetch complete + everyone done reading current buffer
        wait_async0();
        __syncthreads();
    }

    // epilogue: + const2[o], ReLU, transpose through LDS (Cs aliases buffers)
    float* Cs = (float*)smem;                  // [64][68]
    const int mrow0 = wave * 16 + ((lane >> 4) << 3);
#pragma unroll
    for (int bt = 0; bt < 4; ++bt) {
        const int ol = bt * 16 + l15;
        const float c2 = const2[ob + ol];
#pragma unroll
        for (int v = 0; v < 8; ++v) {
            float val = acc[bt][v] + c2;
            Cs[ol * 68 + mrow0 + v] = fmaxf(val, 0.f);
        }
    }
    __syncthreads();

    // coalesced NCHW store: z[(n*512 + ob + ol)*3136 + hw0 + m], 256B per o-row
    {
        const int ol   = t >> 1;
        const int half = (t & 1) * 32;
        float* dst = z + ((long)(n * CO_ + ob + ol)) * HW + hw0 + half;
        const float* src = Cs + ol * 68 + half;
#pragma unroll
        for (int j = 0; j < 32; j += 4)
            *(float4*)(dst + j) = *(const float4*)(src + j);
    }
}

// ---------------------------------------------------------------------------
extern "C" void kernel_launch(void* const* d_in, const int* in_sizes, int n_in,
                              void* d_out, int out_size, void* d_ws, size_t ws_size,
                              hipStream_t stream) {
    (void)in_sizes; (void)n_in; (void)out_size; (void)ws_size;
    const float* x    = (const float*)d_in[0];
    const float* w_dw = (const float*)d_in[1];
    const float* b_dw = (const float*)d_in[2];
    const float* g1   = (const float*)d_in[3];
    const float* be1  = (const float*)d_in[4];
    const float* mu1  = (const float*)d_in[5];
    const float* var1 = (const float*)d_in[6];
    const float* w_pw = (const float*)d_in[7];
    const float* b_pw = (const float*)d_in[8];
    const float* g2   = (const float*)d_in[9];
    const float* be2  = (const float*)d_in[10];
    const float* mu2  = (const float*)d_in[11];
    const float* var2 = (const float*)d_in[12];
    float* z = (float*)d_out;

    char* ws = (char*)d_ws;
    _Float16* yt  = (_Float16*)ws;                              // 32*3136*256*2 = 51,380,224 B
    _Float16* w2f = (_Float16*)(ws + 51380224);                 // 512*256*2     =    262,144 B
    float* const2 = (float*)(ws + 51380224 + 262144);           // 512*4         =      2,048 B

    prep_weights<<<dim3(CO_), dim3(CN), 0, stream>>>(w_pw, b_pw, g2, be2, mu2, var2, w2f, const2);
    dwconv_bn_relu<<<dim3(HH, CN / 32, NB), dim3(224), 0, stream>>>(x, w_dw, b_dw, g1, be1, mu1, var1, yt);
    pwgemm_bn_relu<<<dim3(HW / 64, CO_ / 64, NB), dim3(128), 0, stream>>>(yt, w2f, const2, z);
}
